// WarpAttn_2551210574048
// MI455X (gfx1250) — compile-verified, matrix operand order checked by reading, not run on hardware
//
#include <hip/hip_runtime.h>
#include <hip/hip_bf16.h>

typedef __attribute__((ext_vector_type(16))) _Float16 v16h;
typedef __attribute__((ext_vector_type(8)))  _Float16 v8h;
typedef __attribute__((ext_vector_type(8)))  float    v8f;

#define C_DIM   320
#define HEADS   10
#define HDIM    32
#define E_TOK   512
#define MP_TOK  1536
#define BF      2
#define NEGBIG  (-1000000000.0f)
#define PI_F    3.14159265358979323846f
#define PERS_OUT_SZ 983040
#define ATTN_SCALE 0.17677669529663687f   // 1/sqrt(32)

// ---------------- workspace layout (float offsets) ----------------
#define OFF_ELL     0u          // 512*2
#define OFF_PLL     1024u       // 1536*2
#define OFF_PDIRS   4096u       // 1536*3
#define OFF_EQUIPE  8704u       // 512*320
#define OFF_PPE     172544u     // 1536*320
#define OFF_BIAS1   664064u     // 512*1536
#define OFF_BIAS2   1450496u    // 1536*512
#define OFF_Q1      2236928u    // 2*512*320
#define OFF_KV1     2564608u    // 2*1536*320
#define OFF_Q2      3547648u    // 2*1536*320
#define OFF_KV2     4530688u    // 2*512*320
#define OFF_HQ      4858368u    // 2*1536*320
#define OFF_QB      5841408u
#define OFF_KB      6824448u
#define OFF_VB      7807488u    // V stored TRANSPOSED: [b][h*32+d][Tk]
#define OFF_OB      8790528u
#define OFF_X2      9773568u
#define OFF_G       10756608u   // 2*1536*2560

// ---------------- WMMA helpers ----------------
__device__ __forceinline__ v8f wmma16(v16h a, v16h b, v8f c) {
  return __builtin_amdgcn_wmma_f32_16x16x32_f16(false, a, false, b, (short)0, c, false, false);
}

// K index covered by fragment half t (0..15) for lane-group g (0/1):
//   k = 8*g + (t<8 ? t : t+8)
// A frag: lane%16 = M row.  B frag: lane%16 = N column (matches C/D lane=N).
__device__ __forceinline__ v16h frag_row_f32(const float* __restrict__ p, int g8) {
  const float4 f0 = *(const float4*)(p + g8);
  const float4 f1 = *(const float4*)(p + g8 + 4);
  const float4 f2 = *(const float4*)(p + 16 + g8);
  const float4 f3 = *(const float4*)(p + 20 + g8);
  v16h r;
  r[0]=(_Float16)f0.x;  r[1]=(_Float16)f0.y;  r[2]=(_Float16)f0.z;  r[3]=(_Float16)f0.w;
  r[4]=(_Float16)f1.x;  r[5]=(_Float16)f1.y;  r[6]=(_Float16)f1.z;  r[7]=(_Float16)f1.w;
  r[8]=(_Float16)f2.x;  r[9]=(_Float16)f2.y;  r[10]=(_Float16)f2.z; r[11]=(_Float16)f2.w;
  r[12]=(_Float16)f3.x; r[13]=(_Float16)f3.y; r[14]=(_Float16)f3.z; r[15]=(_Float16)f3.w;
  return r;
}

__device__ __forceinline__ v16h frag_lds(const _Float16* __restrict__ rowp, int g8) {
  v8h lo = *(const v8h*)(rowp + g8);
  v8h hi = *(const v8h*)(rowp + 16 + g8);
  v16h r;
#pragma unroll
  for (int i = 0; i < 8; ++i) { r[i] = lo[i]; r[8+i] = hi[i]; }
  return r;
}

// ---------------- generic WMMA GEMM (templated epilogue) ----------------
// out[M,N] = A[M,K](lda) @ Bw[K,N] (+bias[N]) (+res[M,N]) with optional scatter
// EPI: 0 plain, 1 +bias, 2 +bias+res, 3 +bias+res scatter PERS,
//      4 +bias+res scatter EQUI, 5 plain, store transposed [b][col][tok] (aux=T)
// Block tile 128x64, 8 waves, each wave owns a 32x32 sub-tile (4 WMMA / K-step).
template <int EPI>
__global__ __launch_bounds__(256) void gemm_kernel(
    const float* __restrict__ A, int lda,
    const float* __restrict__ Bw,
    const float* __restrict__ bias,
    const float* __restrict__ res,
    float* __restrict__ out,
    int M, int N, int K, int aux) {
  constexpr bool HAS_BIAS = (EPI >= 1 && EPI <= 4);
  constexpr bool HAS_RES  = (EPI >= 2 && EPI <= 4);

  __shared__ __align__(16) _Float16 As[128 * 40];
  __shared__ __align__(16) _Float16 Bs[64 * 40];

  const int tid  = threadIdx.x;
  const int lane = tid & 31;
  const int ln16 = lane & 15;
  const int lg   = lane >> 4;     // lane group 0/1
  const int g8   = lg * 8;
  const int w    = tid >> 5;      // wave 0..7
  const int wm   = w >> 1;        // 0..3  (32 rows each)
  const int wn   = w & 1;         // 0..1  (32 cols each)
  const int Mb   = blockIdx.y * 128;
  const int Nb   = blockIdx.x * 64;

  const int arow = tid >> 1;      // 0..127
  const int acg  = tid & 1;       // halves of 32-wide row: cols acg*16
  const int bk   = tid >> 3;      // 0..31
  const int bng  = tid & 7;       // cols bng*8

  v8f acc00 = {0,0,0,0,0,0,0,0};
  v8f acc01 = {0,0,0,0,0,0,0,0};
  v8f acc10 = {0,0,0,0,0,0,0,0};
  v8f acc11 = {0,0,0,0,0,0,0,0};

  for (int k0 = 0; k0 < K; k0 += 32) {
    // stage A tile 128x32 (f32 -> f16): 16 floats / thread
    {
      const float* ap = A + (size_t)(Mb + arow) * lda + k0 + acg * 16;
      if (k0 + 32 < K) __builtin_prefetch(ap + 32, 0, 1);
      float4 a0 = *(const float4*)(ap);
      float4 a1 = *(const float4*)(ap + 4);
      float4 a2 = *(const float4*)(ap + 8);
      float4 a3 = *(const float4*)(ap + 12);
      v8h lo, hi;
      lo[0]=(_Float16)a0.x; lo[1]=(_Float16)a0.y; lo[2]=(_Float16)a0.z; lo[3]=(_Float16)a0.w;
      lo[4]=(_Float16)a1.x; lo[5]=(_Float16)a1.y; lo[6]=(_Float16)a1.z; lo[7]=(_Float16)a1.w;
      hi[0]=(_Float16)a2.x; hi[1]=(_Float16)a2.y; hi[2]=(_Float16)a2.z; hi[3]=(_Float16)a2.w;
      hi[4]=(_Float16)a3.x; hi[5]=(_Float16)a3.y; hi[6]=(_Float16)a3.z; hi[7]=(_Float16)a3.w;
      _Float16* dst = As + arow * 40 + acg * 16;
      *(v8h*)(dst)     = lo;
      *(v8h*)(dst + 8) = hi;
    }
    // stage B tile 32x64 transposed -> Bs[n][k]
    {
      const float* bp = Bw + (size_t)(k0 + bk) * N + Nb + bng * 8;
      if (k0 + 32 < K) __builtin_prefetch(bp + 32 * N, 0, 1);
      float4 b0 = *(const float4*)(bp);
      float4 b1 = *(const float4*)(bp + 4);
      float bvals[8] = {b0.x, b0.y, b0.z, b0.w, b1.x, b1.y, b1.z, b1.w};
#pragma unroll
      for (int j = 0; j < 8; ++j)
        Bs[(bng * 8 + j) * 40 + bk] = (_Float16)bvals[j];
    }
    __syncthreads();

    v16h af0 = frag_lds(As + (wm * 32 + ln16) * 40, g8);
    v16h af1 = frag_lds(As + (wm * 32 + 16 + ln16) * 40, g8);
    v16h bf0 = frag_lds(Bs + (wn * 32 + ln16) * 40, g8);
    v16h bf1 = frag_lds(Bs + (wn * 32 + 16 + ln16) * 40, g8);
    acc00 = wmma16(af0, bf0, acc00);
    acc01 = wmma16(af0, bf1, acc01);
    acc10 = wmma16(af1, bf0, acc10);
    acc11 = wmma16(af1, bf1, acc11);
    __syncthreads();
  }

  // epilogue: C/D layout: lane = N (%16), M = r + 8*lg
#pragma unroll
  for (int sm = 0; sm < 2; ++sm) {
#pragma unroll
    for (int sn = 0; sn < 2; ++sn) {
      const v8f& acc = sm ? (sn ? acc11 : acc10) : (sn ? acc01 : acc00);
      int col = Nb + wn * 32 + sn * 16 + ln16;
      float bv = 0.0f;
      if (HAS_BIAS) bv = bias[col];
#pragma unroll
      for (int r = 0; r < 8; ++r) {
        int row = Mb + wm * 32 + sm * 16 + (lg << 3) + r;
        float val = acc[r] + bv;
        if (HAS_RES) val += res[(size_t)row * N + col];
        if (EPI <= 2) {
          out[(size_t)row * N + col] = val;
        } else if (EPI == 3) {             // pers scatter, Tq = 1536
          int f   = row / 1536;
          int rem = row - f * 1536;
          int mi  = rem >> 8;
          int p   = rem & 255;
          int y   = p >> 4, xx = p & 15;
          out[((((size_t)mi * 320 + col) * 2 + f) * 16 + y) * 16 + xx] = val;
        } else if (EPI == 4) {              // equi scatter, Tq = 512
          int f = row / 512;
          int e = row - f * 512;
          int y = e >> 5, xx = e & 31;
          out[(size_t)PERS_OUT_SZ + (((size_t)col * 2 + f) * 16 + y) * 32 + xx] = val;
        } else {                            // EPI == 5: transposed [b][col][tok], aux = T
          int bb  = row / aux;
          int tok = row - bb * aux;
          out[((size_t)bb * C_DIM + col) * aux + tok] = val;
        }
      }
    }
  }
}

// ---------------- flash attention (one wave per 16-query tile) ----------------
// q,k: [bf*T, 320] head h at cols h*32..   vT: [b][h*32+d][Tk]
// bias: [Tq, Tk]   o: [bf*Tq, 320]
__global__ __launch_bounds__(32) void attn_kernel(
    const float* __restrict__ q, const float* __restrict__ k,
    const float* __restrict__ vT, const float* __restrict__ bias,
    float* __restrict__ o, int Tq, int Tk) {
  const int lane = threadIdx.x;
  const int ln16 = lane & 15;
  const int lg   = lane >> 4;
  const int g8   = lg * 8;
  const int b    = blockIdx.y / HEADS;
  const int h    = blockIdx.y % HEADS;
  const int qg   = blockIdx.x * 16 + ln16;   // this lane's query row

  // Q as B-fragment: lane = query, halves = hdim per k-mapping
  v16h qb = frag_row_f32(q + ((size_t)(b * Tq + qg)) * C_DIM + h * HDIM, g8);

  // per-lane V^T row pointers (contiguous in kc thanks to transposed store)
  const float* v0row = vT + ((size_t)b * C_DIM + h * HDIM + ln16)      * Tk;
  const float* v1row = vT + ((size_t)b * C_DIM + h * HDIM + 16 + ln16) * Tk;

  v8f o0 = {0,0,0,0,0,0,0,0};   // o^T rows d = g8 + r
  v8f o1 = {0,0,0,0,0,0,0,0};   // o^T rows d = 16 + g8 + r
  float m_run = -__builtin_inff();
  float l_run = 0.0f;

  for (int kb0 = 0; kb0 < Tk; kb0 += 32) {
    // K rows as A-fragments (M = k index)
    v16h a0 = frag_row_f32(k + (size_t)(b * Tk + kb0 + ln16)      * C_DIM + h * HDIM, g8);
    v16h a1 = frag_row_f32(k + (size_t)(b * Tk + kb0 + 16 + ln16) * C_DIM + h * HDIM, g8);

    v8f z = {0,0,0,0,0,0,0,0};
    v8f s0 = wmma16(a0, qb, z);   // s'[kc, q], kc = r + 8*lg        (+kb0)
    v8f s1 = wmma16(a1, qb, z);   // kc = 16 + r + 8*lg

    // scale + mask bias (bias row = query, contiguous in kc)
    const float* bp = bias + (size_t)qg * Tk + kb0;
    float4 bq0 = *(const float4*)(bp + g8);
    float4 bq1 = *(const float4*)(bp + g8 + 4);
    float4 bq2 = *(const float4*)(bp + 16 + g8);
    float4 bq3 = *(const float4*)(bp + 20 + g8);
    float bl0[8] = {bq0.x,bq0.y,bq0.z,bq0.w,bq1.x,bq1.y,bq1.z,bq1.w};
    float bl1[8] = {bq2.x,bq2.y,bq2.z,bq2.w,bq3.x,bq3.y,bq3.z,bq3.w};
#pragma unroll
    for (int r = 0; r < 8; ++r) {
      s0[r] = s0[r] * ATTN_SCALE + bl0[r];
      s1[r] = s1[r] * ATTN_SCALE + bl1[r];
    }

    // online softmax over kc (8 regs x 2 tiles in-lane, + partner half-wave)
    float mloc = -__builtin_inff();
#pragma unroll
    for (int r = 0; r < 8; ++r) mloc = fmaxf(mloc, fmaxf(s0[r], s1[r]));
    mloc = fmaxf(mloc, __shfl_xor(mloc, 16, 32));
    float mnew = fmaxf(m_run, mloc);
    float corr = __expf(m_run - mnew);
    float lloc = 0.0f;
#pragma unroll
    for (int r = 0; r < 8; ++r) {
      s0[r] = __expf(s0[r] - mnew);
      s1[r] = __expf(s1[r] - mnew);
      lloc += s0[r] + s1[r];
    }
    lloc += __shfl_xor(lloc, 16, 32);
    l_run = l_run * corr + lloc;
    m_run = mnew;
#pragma unroll
    for (int r = 0; r < 8; ++r) { o0[r] *= corr; o1[r] *= corr; }

    // probs D-frag -> B-frag for o^T GEMM: zero cross-lane movement
    v16h pb;
#pragma unroll
    for (int t = 0; t < 16; ++t) {
      int vv = t >> 1, ii = t & 1;
      int r  = ((vv & 3) << 1) | ii;
      float pv = (vv >= 4) ? s1[r] : s0[r];
      pb[t] = (_Float16)pv;
    }

    // V^T as A-fragments: contiguous float4 loads from transposed V
    v16h va0 = frag_row_f32(v0row + kb0, g8);
    v16h va1 = frag_row_f32(v1row + kb0, g8);
    o0 = wmma16(va0, pb, o0);
    o1 = wmma16(va1, pb, o1);
  }

  float inv = 1.0f / l_run;
  float* op = o + ((size_t)(b * Tq + qg)) * C_DIM + h * HDIM;
  float4 w0 = {o0[0]*inv, o0[1]*inv, o0[2]*inv, o0[3]*inv};
  float4 w1 = {o0[4]*inv, o0[5]*inv, o0[6]*inv, o0[7]*inv};
  float4 w2 = {o1[0]*inv, o1[1]*inv, o1[2]*inv, o1[3]*inv};
  float4 w3 = {o1[4]*inv, o1[5]*inv, o1[6]*inv, o1[7]*inv};
  *(float4*)(op + g8)      = w0;
  *(float4*)(op + g8 + 4)  = w1;
  *(float4*)(op + 16 + g8) = w2;
  *(float4*)(op + 20 + g8) = w3;
}

// ---------------- LayerNorm (+optional query PE) ----------------
__global__ __launch_bounds__(32) void ln_kernel(
    const float* __restrict__ x, const float* __restrict__ w,
    const float* __restrict__ b, const float* __restrict__ qpe, int Tq,
    float* __restrict__ out) {
  const int row  = blockIdx.x;
  const int lane = threadIdx.x;
  const float* xr = x + (size_t)row * C_DIM;
  float vals[10];
  float s = 0.f, ss = 0.f;
#pragma unroll
  for (int j = 0; j < 10; ++j) {
    float vv = xr[lane + 32 * j];
    vals[j] = vv; s += vv; ss += vv * vv;
  }
#pragma unroll
  for (int off = 16; off > 0; off >>= 1) {
    s  += __shfl_xor(s,  off, 32);
    ss += __shfl_xor(ss, off, 32);
  }
  float mu  = s * (1.0f / C_DIM);
  float var = ss * (1.0f / C_DIM) - mu * mu;
  float rstd = rsqrtf(var + 1e-5f);
  int tok = row % Tq;
#pragma unroll
  for (int j = 0; j < 10; ++j) {
    int c = lane + 32 * j;
    float hh = (vals[j] - mu) * rstd * w[c] + b[c];
    if (qpe) hh += qpe[(size_t)tok * C_DIM + c];
    out[(size_t)row * C_DIM + c] = hh;
  }
}

// ---------------- gated-GELU activation (in-place on g) ----------------
__global__ void act_kernel(float* __restrict__ g, int total) {
  int idx = blockIdx.x * blockDim.x + threadIdx.x;
  if (idx >= total) return;
  int r = idx / 1280, j = idx - r * 1280;
  float hid  = g[(size_t)r * 2560 + j];
  float gate = g[(size_t)r * 2560 + 1280 + j];
  float g3 = gate * gate * gate;
  float ge = 0.5f * gate * (1.0f + tanhf(0.7978845608028654f * (gate + 0.044715f * g3)));
  g[(size_t)r * 2560 + j] = hid * ge;
}

// ---------------- geometry / PE / bias / prep ----------------
__global__ void fill_equi_ll(float* __restrict__ ell) {
  int e = blockIdx.x * blockDim.x + threadIdx.x;
  if (e >= E_TOK) return;
  int y = e >> 5, x = e & 31;
  ell[e * 2 + 0] = (0.5f - (y + 0.5f) / 16.0f) * PI_F;
  ell[e * 2 + 1] = ((x + 0.5f) / 32.0f - 0.5f) * (2.0f * PI_F);
}

__global__ void pers_geom_kernel(const float* __restrict__ cameras,
                                 float* __restrict__ pll, float* __restrict__ pdirs) {
  int p = blockIdx.x * blockDim.x + threadIdx.x;
  if (p >= MP_TOK) return;
  int mi = p >> 8, rem = p & 255, y = rem >> 4, x = rem & 15;
  float fov = cameras[mi * 3 + 0], yaw = cameras[mi * 3 + 1], pit = cameras[mi * 3 + 2];
  float t  = tanf(fov * 0.5f);
  float u  = (x + 0.5f) / 16.0f * 2.0f - 1.0f;
  float vv = 1.0f - (y + 0.5f) / 16.0f * 2.0f;
  float dx = t * u, dy = t * vv, dz = 1.0f;
  float inv = rsqrtf(dx * dx + dy * dy + dz * dz);
  dx *= inv; dy *= inv; dz *= inv;
  float cp = cosf(pit), sp = sinf(pit), cy = cosf(yaw), sy = sinf(yaw);
  float wx = dx, wy = cp * dy - sp * dz, wz = sp * dy + cp * dz;   // Rx
  float ox = cy * wx + sy * wz, oy = wy, oz = -sy * wx + cy * wz;  // Ry
  float cl = fminf(1.0f, fmaxf(-1.0f, oy));
  pll[p * 2 + 0] = asinf(cl);
  pll[p * 2 + 1] = atan2f(ox, oz);
  pdirs[p * 3 + 0] = ox; pdirs[p * 3 + 1] = oy; pdirs[p * 3 + 2] = oz;
}

__global__ void pe_kernel(const float* __restrict__ ll, float* __restrict__ pe, int total) {
  int idx = blockIdx.x * blockDim.x + threadIdx.x;
  if (idx >= total) return;
  int n = idx / C_DIM, c = idx - n * C_DIM;
  int coord = c / 160, j = c - coord * 160;
  float val  = ll[n * 2 + coord];
  float freq = exp2f((float)(j < 80 ? j : j - 80));
  float ang  = val * freq;
  pe[idx] = (j < 80) ? sinf(ang) : cosf(ang);
}

__global__ void bias_kernel(const float* __restrict__ cameras,
                            const float* __restrict__ ell,
                            const float* __restrict__ pdirs,
                            float* __restrict__ bias1, float* __restrict__ bias2) {
  int idx = blockIdx.x * blockDim.x + threadIdx.x;
  if (idx >= E_TOK * MP_TOK) return;
  int e = idx / MP_TOK, p = idx - e * MP_TOK;
  float lat = ell[e * 2 + 0], lon = ell[e * 2 + 1];
  float cl = cosf(lat);
  float ex = cl * sinf(lon), ey = sinf(lat), ez = cl * cosf(lon);
  float dot = pdirs[p*3+0]*ex + pdirs[p*3+1]*ey + pdirs[p*3+2]*ez;
  int mi = p >> 8;
  float th = cosf(cameras[mi * 3 + 0] * 0.6f);
  float bv = (dot > th) ? 0.0f : NEGBIG;
  bias1[(size_t)e * MP_TOK + p] = bv;
  bias2[(size_t)p * E_TOK + e]  = bv;
}

__global__ void prep_q1_kv2(const float* __restrict__ equi_x,
                            const float* __restrict__ equi_pe,
                            float* __restrict__ q1, float* __restrict__ kv2) {
  int idx = blockIdx.x * blockDim.x + threadIdx.x;
  if (idx >= BF * E_TOK * C_DIM) return;
  int r = idx / C_DIM, c = idx - r * C_DIM;
  int f = r / E_TOK, e = r - f * E_TOK;
  int y = e >> 5, x = e & 31;
  float val = equi_x[(((size_t)c * 2 + f) * 16 + y) * 32 + x];
  q1[idx]  = val;
  kv2[idx] = val + equi_pe[(size_t)e * C_DIM + c];
}

__global__ void prep_q2_kv1(const float* __restrict__ pers_x,
                            const float* __restrict__ pers_pe,
                            float* __restrict__ q2, float* __restrict__ kv1) {
  int idx = blockIdx.x * blockDim.x + threadIdx.x;
  if (idx >= BF * MP_TOK * C_DIM) return;
  int r = idx / C_DIM, c = idx - r * C_DIM;
  int f = r / MP_TOK, rem = r - f * MP_TOK;
  int mi = rem >> 8, p = rem & 255, y = p >> 4, x = p & 15;
  float val = pers_x[((((size_t)mi * C_DIM + c) * 2 + f) * 16 + y) * 16 + x];
  q2[idx]  = val;
  kv1[idx] = val + pers_pe[(size_t)rem * C_DIM + c];
}

// ---------------- host side ----------------
static inline void launch_gemm(int epi, dim3 grid, hipStream_t s,
                               const float* A, int lda, const float* Bw,
                               const float* bias, const float* res, float* out,
                               int M, int N, int K, int aux = 0) {
  switch (epi) {
    case 0: gemm_kernel<0><<<grid, 256, 0, s>>>(A, lda, Bw, bias, res, out, M, N, K, aux); break;
    case 1: gemm_kernel<1><<<grid, 256, 0, s>>>(A, lda, Bw, bias, res, out, M, N, K, aux); break;
    case 2: gemm_kernel<2><<<grid, 256, 0, s>>>(A, lda, Bw, bias, res, out, M, N, K, aux); break;
    case 3: gemm_kernel<3><<<grid, 256, 0, s>>>(A, lda, Bw, bias, res, out, M, N, K, aux); break;
    case 4: gemm_kernel<4><<<grid, 256, 0, s>>>(A, lda, Bw, bias, res, out, M, N, K, aux); break;
    default: gemm_kernel<5><<<grid, 256, 0, s>>>(A, lda, Bw, bias, res, out, M, N, K, aux); break;
  }
}

static void run_block(const float* x, const float* ctx, const float* bias,
                      const float* qpe, int Tq, int Tk, int scatter_epi,
                      float* ws, void* const* d_in, float* dout, hipStream_t s) {
  const float* ln1_w = (const float*)d_in[3];
  const float* ln1_b = (const float*)d_in[4];
  const float* w_q   = (const float*)d_in[5];
  const float* w_k   = (const float*)d_in[6];
  const float* w_v   = (const float*)d_in[7];
  const float* w_o   = (const float*)d_in[8];
  const float* b_o   = (const float*)d_in[9];
  const float* ln2_w = (const float*)d_in[10];
  const float* ln2_b = (const float*)d_in[11];
  const float* ff_w1 = (const float*)d_in[12];
  const float* ff_b1 = (const float*)d_in[13];
  const float* ff_w2 = (const float*)d_in[14];
  const float* ff_b2 = (const float*)d_in[15];

  float* hq = ws + OFF_HQ;
  float* qb = ws + OFF_QB;
  float* kb = ws + OFF_KB;
  float* vb = ws + OFF_VB;   // transposed layout [b][c][Tk]
  float* ob = ws + OFF_OB;
  float* x2 = ws + OFF_X2;
  float* g  = ws + OFF_G;

  const int Mq = BF * Tq, Mk = BF * Tk;

  ln_kernel<<<Mq, 32, 0, s>>>(x, ln1_w, ln1_b, qpe, Tq, hq);
  launch_gemm(0, dim3(C_DIM/64, Mq/128), s, hq, C_DIM, w_q, nullptr, nullptr, qb, Mq, C_DIM, C_DIM);
  launch_gemm(0, dim3(C_DIM/64, Mk/128), s, ctx, C_DIM, w_k, nullptr, nullptr, kb, Mk, C_DIM, C_DIM);
  launch_gemm(5, dim3(C_DIM/64, Mk/128), s, ctx, C_DIM, w_v, nullptr, nullptr, vb, Mk, C_DIM, C_DIM, Tk);
  attn_kernel<<<dim3(Tq/16, BF*HEADS), 32, 0, s>>>(qb, kb, vb, bias, ob, Tq, Tk);
  launch_gemm(2, dim3(C_DIM/64, Mq/128), s, ob, C_DIM, w_o, b_o, x, x2, Mq, C_DIM, C_DIM);
  ln_kernel<<<Mq, 32, 0, s>>>(x2, ln2_w, ln2_b, nullptr, Tq, hq);
  launch_gemm(1, dim3(2560/64, Mq/128), s, hq, C_DIM, ff_w1, ff_b1, nullptr, g, Mq, 2560, C_DIM);
  {
    int total = Mq * 1280;
    act_kernel<<<(total + 255)/256, 256, 0, s>>>(g, total);
  }
  launch_gemm(scatter_epi, dim3(C_DIM/64, Mq/128), s, g, 2560, ff_w2, ff_b2, x2, dout, Mq, C_DIM, 1280);
}

extern "C" void kernel_launch(void* const* d_in, const int* in_sizes, int n_in,
                              void* d_out, int out_size, void* d_ws, size_t ws_size,
                              hipStream_t stream) {
  (void)in_sizes; (void)n_in; (void)out_size; (void)ws_size;
  const float* pers_x  = (const float*)d_in[0];
  const float* equi_x  = (const float*)d_in[1];
  const float* cameras = (const float*)d_in[2];
  float* ws  = (float*)d_ws;
  float* out = (float*)d_out;

  // geometry + PE + mask biases
  fill_equi_ll<<<2, 256, 0, stream>>>(ws + OFF_ELL);
  pers_geom_kernel<<<6, 256, 0, stream>>>(cameras, ws + OFF_PLL, ws + OFF_PDIRS);
  pe_kernel<<<(E_TOK*C_DIM + 255)/256, 256, 0, stream>>>(ws + OFF_ELL, ws + OFF_EQUIPE, E_TOK*C_DIM);
  pe_kernel<<<(MP_TOK*C_DIM + 255)/256, 256, 0, stream>>>(ws + OFF_PLL, ws + OFF_PPE, MP_TOK*C_DIM);
  bias_kernel<<<(E_TOK*MP_TOK + 255)/256, 256, 0, stream>>>(cameras, ws + OFF_ELL, ws + OFF_PDIRS,
                                                            ws + OFF_BIAS1, ws + OFF_BIAS2);
  // activation prep (token-major layouts)
  prep_q1_kv2<<<(BF*E_TOK*C_DIM + 255)/256, 256, 0, stream>>>(equi_x, ws + OFF_EQUIPE,
                                                              ws + OFF_Q1, ws + OFF_KV2);
  prep_q2_kv1<<<(BF*MP_TOK*C_DIM + 255)/256, 256, 0, stream>>>(pers_x, ws + OFF_PPE,
                                                               ws + OFF_Q2, ws + OFF_KV1);

  // block 1: equi queries attend to perspective kv  -> equi_out (scatter epi 4)
  run_block(ws + OFF_Q1, ws + OFF_KV1, ws + OFF_BIAS1, ws + OFF_EQUIPE,
            E_TOK, MP_TOK, 4, ws, d_in, out, stream);
  // block 2: perspective queries attend to equi kv  -> pers_out (scatter epi 3)
  run_block(ws + OFF_Q2, ws + OFF_KV2, ws + OFF_BIAS2, ws + OFF_PPE,
            MP_TOK, E_TOK, 3, ws, d_in, out, stream);
}
